// Model_30846455120276
// MI455X (gfx1250) — matrix operand order repaired and checked by reading, not compiled, not run on hardware
//
#include <hip/hip_runtime.h>
#include <math.h>

// ---------------------------------------------------------------------------
// Problem constants (from the reference)
// ---------------------------------------------------------------------------
#define KSIZE   337      // conv kernel length == contraction dim K
#define OUT_CH  300      // conv output channels == GEMM N
#define W_WORDS 128      // words per sample == GEMM M
#define BATCH   512
#define NPAD    304      // 19 * 16  (N padded to WMMA tiles)
#define NT      19       // N tiles of 16
#define KC      11       // ceil(337 / 32) K-chunks of 32
#define BPB     4        // batches processed per block (amortize LDS staging)

// LDS layout (total 320,768 B <= 320 KB WGP LDS):
//   [0)        f16 weights [NPAD][360]   218,880 B  (stride 360 halves = 180
//              dwords; 180 % 64 = 52 -> 16-lane b128 reads hit 64 distinct
//              banks, conflict-free)
//   [WBYTES)   per-wave f16 A tiles, 8 x [16][360]  92,160 B
//   [ABYTES)   wave max buffer [8][NPAD] f32         9,728 B
#define LDS_WSTRIDE 360
#define LDS_WBYTES  (NPAD * LDS_WSTRIDE * 2)              // 218,880
#define LDS_ATILE   (16 * LDS_WSTRIDE * 2)                // 11,520 per wave
#define LDS_ABASE   LDS_WBYTES
#define LDS_MAXBASE (LDS_WBYTES + 8 * LDS_ATILE)          // 311,040
#define LDS_TOTAL   (LDS_MAXBASE + 8 * NPAD * 4)          // 320,768

typedef _Float16 half16 __attribute__((ext_vector_type(16)));
typedef _Float16 half8  __attribute__((ext_vector_type(8)));
typedef float    floatx8 __attribute__((ext_vector_type(8)));

// Assemble a 16x32 f16 operand fragment from two 16B-aligned LDS half8 loads.
__device__ __forceinline__ half16 frag_ld(const _Float16* p0, const _Float16* p1) {
  half8 lo = *(const half8*)p0;      // ds_load_b128
  half8 hi = *(const half8*)p1;      // ds_load_b128
  half16 r;
#pragma unroll
  for (int i = 0; i < 8; ++i) { r[i] = lo[i]; r[8 + i] = hi[i]; }
  return r;
}

// Per-lane relu-max epilogue: fold 8 rows, then the two M halves, stash in LDS.
__device__ __forceinline__ void tile_max(floatx8 acc, int n0, int wave, int l,
                                         int half, float* maxpart) {
  float mx = acc[0];
#pragma unroll
  for (int r = 1; r < 8; ++r) mx = fmaxf(mx, acc[r]);
  mx = fmaxf(mx, __shfl_xor(mx, 16, 32));
  if (half == 0) maxpart[wave * NPAD + n0 + l] = mx;
}

// ---------------------------------------------------------------------------
// Stage 1: per-sample conv-as-GEMM + relu + max over words, via WMMA f16.
//   grid: 256 blocks  (input in {0,1}) x (128 batch groups of BPB=4)
//   block: 256 threads = 8 wave32 waves; wave w owns M rows [16w, 16w+16)
// ---------------------------------------------------------------------------
__global__ __launch_bounds__(256, 1)
void encode_kernel(const float* __restrict__ x1, const float* __restrict__ x2,
                   const float* __restrict__ conv_w, const float* __restrict__ conv_b,
                   float* __restrict__ vbuf /* [2][512][300] */) {
  extern __shared__ char smem[];
  _Float16* wlds    = (_Float16*)smem;
  float*    maxpart = (float*)(smem + LDS_MAXBASE);

  const int tid  = threadIdx.x;
  const int lane = tid & 31;
  const int wave = tid >> 5;
  const int l    = lane & 15;     // position within 16-lane half
  const int half = lane >> 4;     // 0: lanes 0-15, 1: lanes 16-31

  const int input  = blockIdx.x >> 7;       // 0 -> x1, 1 -> x2
  const int bgroup = blockIdx.x & 127;
  const float* xin = input ? x2 : x1;

  _Float16* atile = (_Float16*)(smem + LDS_ABASE + wave * LDS_ATILE);

  // ---- stage conv_w fp32 -> f16 into LDS, zero-padded to [304][360] ----
  for (int t = tid; t < NPAD * 45; t += 256) {
    int row = t / 45;
    int g   = t - row * 45;
    int k   = g * 8;
    half8 v;
#pragma unroll
    for (int i = 0; i < 8; ++i) {
      float f = 0.0f;
      if (row < OUT_CH && (k + i) < KSIZE) f = conv_w[row * KSIZE + k + i];
      v[i] = (_Float16)f;
    }
    *(half8*)(&wlds[row * LDS_WSTRIDE + k]) = v;   // ds_store_b128
  }

  // ---- zero the K-pad columns of this wave's A tile (persists, per-wave) ----
  for (int t = lane; t < 16 * 23; t += 32) {
    int r = t / 23;
    int c = KSIZE + (t - r * 23);                  // cols 337..359
    atile[r * LDS_WSTRIDE + c] = (_Float16)0.0f;
  }
  __syncthreads();

  const int m0 = wave * 16;

  for (int ib = 0; ib < BPB; ++ib) {
    const int b = bgroup * BPB + ib;

    // ---- coalesced staging of the 16x337 A slab (contiguous in memory) ----
    // lane walks consecutive dwords: each global_load_b32 covers one 128B line.
    const float* slab = xin + ((size_t)b * W_WORDS + m0) * KSIZE;
    {
      int row = 0, col = lane;
      for (int idx = lane; idx < 16 * KSIZE; idx += 32) {
        atile[row * LDS_WSTRIDE + col] = (_Float16)slab[idx];
        col += 32;
        if (col >= KSIZE) { col -= KSIZE; ++row; }
      }
    }

    // prefetch next batch's slab into cache while we compute (CDNA5 prefetch)
    if (ib + 1 < BPB) {
      const float* nslab = slab + (size_t)W_WORDS * KSIZE;
      for (int p = lane; p < 169; p += 32)
        __builtin_prefetch(nslab + p * 32, 0, 3);  // global_prefetch_b8
    }

    // ---- build this wave's A fragments from LDS (same-wave DS is ordered) --
    // ISA 16-bit A 16x32 layout: lane<16 holds K k0+0..7 then k0+16..23,
    // lane>=16 holds K k0+8..15 then k0+24..31 (elements 0..7 / 8..15).
    const _Float16* arow = atile + l * LDS_WSTRIDE + half * 8;
    half16 afrag[KC];
#pragma unroll
    for (int kc = 0; kc < KC; ++kc) {
      const int k0 = kc * 32;
      afrag[kc] = frag_ld(arow + k0, arow + k0 + 16);
    }

    // ---- GEMM: two N tiles per pass -> two independent WMMA chains ----
    // B 32x16 layout: lane = N column; lanes 0-15 hold K 0..7,16..23 of the chunk,
    // lanes 16-31 hold K 8..15,24..31 -> two 16B reads per lane.
    for (int ntp = 0; ntp < 9; ++ntp) {
      const int n0 = ntp * 32;
      const _Float16* wrow0 = &wlds[(n0 + l) * LDS_WSTRIDE + half * 8];
      const _Float16* wrow1 = wrow0 + 16 * LDS_WSTRIDE;
      floatx8 acc0 = {}, acc1 = {};
#pragma unroll
      for (int kc = 0; kc < KC; ++kc) {
        const int k0 = kc * 32;
        half16 b0 = frag_ld(wrow0 + k0, wrow0 + k0 + 16);
        half16 b1 = frag_ld(wrow1 + k0, wrow1 + k0 + 16);
        acc0 = __builtin_amdgcn_wmma_f32_16x16x32_f16(
            false, afrag[kc], false, b0, (short)0, acc0, false, false);
        acc1 = __builtin_amdgcn_wmma_f32_16x16x32_f16(
            false, afrag[kc], false, b1, (short)0, acc1, false, false);
      }
      tile_max(acc0, n0,      wave, l, half, maxpart);
      tile_max(acc1, n0 + 16, wave, l, half, maxpart);
    }
    {   // tail N tile (nt = 18)
      const int n0 = 18 * 16;
      const _Float16* wrow = &wlds[(n0 + l) * LDS_WSTRIDE + half * 8];
      floatx8 acc = {};
#pragma unroll
      for (int kc = 0; kc < KC; ++kc) {
        const int k0 = kc * 32;
        half16 bf = frag_ld(wrow + k0, wrow + k0 + 16);
        acc = __builtin_amdgcn_wmma_f32_16x16x32_f16(
            false, afrag[kc], false, bf, (short)0, acc, false, false);
      }
      tile_max(acc, n0, wave, l, half, maxpart);
    }
    __syncthreads();

    // ---- reduce over the 8 waves (M tiles), apply bias + relu, store ----
    for (int n = tid; n < NPAD; n += 256) {
      float mx = maxpart[n];
#pragma unroll
      for (int w = 1; w < 8; ++w) mx = fmaxf(mx, maxpart[w * NPAD + n]);
      if (n < OUT_CH)
        vbuf[(size_t)(input * BATCH + b) * OUT_CH + n] =
            fmaxf(mx + conv_b[n], 0.0f);
    }
    __syncthreads();
  }
}

// ---------------------------------------------------------------------------
// Stage 2: tiny FC head  (feat = [v1-v2, v1*v2]; tanh(fc1); fc2) — ~184 MFLOP
//   grid: 512 blocks (one per batch row), 320 threads (10 waves)
// ---------------------------------------------------------------------------
__global__ __launch_bounds__(320, 1)
void head_kernel(const float* __restrict__ vbuf,
                 const float* __restrict__ fc1_w, const float* __restrict__ fc1_b,
                 const float* __restrict__ fc2_w, const float* __restrict__ fc2_b,
                 float* __restrict__ out) {
  __shared__ float feat[600];
  __shared__ float red[320];
  const int b = blockIdx.x;
  const int tid = threadIdx.x;

  const float* v1 = vbuf + (size_t)b * OUT_CH;
  const float* v2 = vbuf + (size_t)(BATCH + b) * OUT_CH;
  for (int j = tid; j < OUT_CH; j += 320) {
    float a = v1[j], c = v2[j];
    feat[j]          = a - c;
    feat[OUT_CH + j] = a * c;
  }
  __syncthreads();

  float p = 0.0f;
  if (tid < OUT_CH) {
    const float* wr = fc1_w + (size_t)tid * 600;
    float s = fc1_b[tid];
#pragma unroll 4
    for (int j = 0; j < 600; ++j) s = fmaf(wr[j], feat[j], s);
    p = tanhf(s) * fc2_w[tid];
  }
  red[tid] = p;
  __syncthreads();
  if (tid < 160) red[tid] += red[tid + 160]; __syncthreads();
  if (tid < 80)  red[tid] += red[tid + 80];  __syncthreads();
  if (tid < 40)  red[tid] += red[tid + 40];  __syncthreads();
  if (tid < 20)  red[tid] += red[tid + 20];  __syncthreads();
  if (tid < 10)  red[tid] += red[tid + 10];  __syncthreads();
  if (tid < 5)   red[tid] += red[tid + 5];   __syncthreads();
  if (tid == 0)
    out[b] = red[0] + red[1] + red[2] + red[3] + red[4] + fc2_b[0];
}

// ---------------------------------------------------------------------------
extern "C" void kernel_launch(void* const* d_in, const int* in_sizes, int n_in,
                              void* d_out, int out_size, void* d_ws, size_t ws_size,
                              hipStream_t stream) {
  const float* x1     = (const float*)d_in[0];
  const float* x2     = (const float*)d_in[1];
  const float* conv_w = (const float*)d_in[2];
  const float* conv_b = (const float*)d_in[3];
  const float* fc1_w  = (const float*)d_in[4];
  const float* fc1_b  = (const float*)d_in[5];
  const float* fc2_w  = (const float*)d_in[6];
  const float* fc2_b  = (const float*)d_in[7];

  float* vbuf = (float*)d_ws;        // [2][512][300] f32 = 1.23 MB scratch
  float* out  = (float*)d_out;       // [512]

  // encode_kernel uses 320,768 B of dynamic LDS (CDNA5 WGP has 320 KB).
  (void)hipFuncSetAttribute((const void*)encode_kernel,
                            hipFuncAttributeMaxDynamicSharedMemorySize,
                            LDS_TOTAL);

  encode_kernel<<<dim3(256), dim3(256), LDS_TOTAL, stream>>>(
      x1, x2, conv_w, conv_b, vbuf);
  head_kernel<<<dim3(512), dim3(320), 0, stream>>>(
      vbuf, fc1_w, fc1_b, fc2_w, fc2_b, out);
}
